// GCN_83073257439785
// MI455X (gfx1250) — matrix operand order repaired and checked by reading, not compiled, not run on hardware
//
#include <hip/hip_runtime.h>
#include <hip/hip_bf16.h>
#include <math.h>

// ---- CDNA5 WMMA vector types -------------------------------------------------
typedef __attribute__((ext_vector_type(16))) __bf16 v16bf;
typedef __attribute__((ext_vector_type(8)))  float  v8f;

#define DFEAT 64

__device__ __forceinline__ __bf16 f2bf(float f) { return (__bf16)f; }  // RNE

// ---- utility fill ------------------------------------------------------------
__global__ void fill_f32(float* __restrict__ p, float v, long long n) {
    long long i = (long long)blockIdx.x * blockDim.x + threadIdx.x;
    if (i < n) p[i] = v;
}

// ---- degree / normalization --------------------------------------------------
__global__ void deg_accum(const int* __restrict__ dst, float* __restrict__ deg, int E) {
    int e = blockIdx.x * blockDim.x + threadIdx.x;
    if (e < E) atomicAdd(&deg[dst[e]], 1.0f);
}

__global__ void dinv_k(float* __restrict__ d, int n) {
    int i = blockIdx.x * blockDim.x + threadIdx.x;
    if (i < n) { float v = d[i]; d[i] = (v > 0.0f) ? rsqrtf(v) : 0.0f; }
}

__global__ void norm_k(const int* __restrict__ src, const int* __restrict__ dst,
                       const float* __restrict__ dinv, float* __restrict__ norm, int E) {
    int e = blockIdx.x * blockDim.x + threadIdx.x;
    if (e < E) norm[e] = dinv[src[e]] * dinv[dst[e]];
}

// ---- W -> bf16, transposed: WT[c*K + k] = bf16(W[k*64 + c]) ------------------
__global__ void wt_conv(const float* __restrict__ W, __bf16* __restrict__ WT, int K) {
    int i = blockIdx.x * blockDim.x + threadIdx.x;
    if (i >= DFEAT * K) return;
    int c = i / K, k = i - c * K;
    WT[i] = f2bf(W[(size_t)k * DFEAT + c]);
}

// ---- WMMA GEMM:  out[N,64] = A[N,K] * W[K,64]  (bf16 inputs, f32 accum) ------
// 256 threads = 8 waves; wave w owns rows m0..m0+15; 4 column tiles of 16.
// A fragment: two contiguous 8-float segments per lane -> 4x global_load_b128.
// B fragment: contiguous 32B bf16 run from transposed WT -> 2x global_load_b128.
// OOB rows: pointer clamped (in-bounds), result rows simply never stored.
template <int K>
__global__ void gemm_wmma_bf16(const float* __restrict__ A, const __bf16* __restrict__ WT,
                               float* __restrict__ out, int nrows) {
    int lane = threadIdx.x & 31;
    int lr   = lane & 15;
    int hi   = (lane >> 4) & 1;
    int m0   = blockIdx.x * 128 + (threadIdx.x >> 5) * 16;

    int r  = m0 + lr;
    int rc = (r < nrows) ? r : (nrows - 1);          // clamp: no OOB, rows discarded
    const float* arow = A + (size_t)rc * K;
    int aoff  = hi ? 16 : 0;   // A: high half-wave holds K 16..31
    int koffB = hi ? 16 : 0;   // B: high half-wave holds K rows 16..31

    v8f acc[4] = {};

#pragma unroll
    for (int k0 = 0; k0 < K; k0 += 32) {
        const float4* s0 = (const float4*)(arow + k0 + aoff);
        const float4* s1 = (const float4*)(arow + k0 + aoff + 8);
        float4 p0 = s0[0], p1 = s0[1];
        float4 q0 = s1[0], q1 = s1[1];
        v16bf af;
        af[0]  = f2bf(p0.x); af[1]  = f2bf(p0.y); af[2]  = f2bf(p0.z); af[3]  = f2bf(p0.w);
        af[4]  = f2bf(p1.x); af[5]  = f2bf(p1.y); af[6]  = f2bf(p1.z); af[7]  = f2bf(p1.w);
        af[8]  = f2bf(q0.x); af[9]  = f2bf(q0.y); af[10] = f2bf(q0.z); af[11] = f2bf(q0.w);
        af[12] = f2bf(q1.x); af[13] = f2bf(q1.y); af[14] = f2bf(q1.z); af[15] = f2bf(q1.w);

#pragma unroll
        for (int nt = 0; nt < 4; ++nt) {
            v16bf bf = *(const v16bf*)(WT + (size_t)(nt * 16 + lr) * K + k0 + koffB);
            acc[nt] = __builtin_amdgcn_wmma_f32_16x16x32_bf16(
                false, af, false, bf, (short)0, acc[nt], false, false);
        }
    }

    int rowhi = hi ? 8 : 0;  // C/D layout: VGPR rr -> M=rr (lanes 0-15) / M=rr+8
#pragma unroll
    for (int nt = 0; nt < 4; ++nt) {
        int c = nt * 16 + lr;
#pragma unroll
        for (int rr = 0; rr < 8; ++rr) {
            int row = m0 + rowhi + rr;
            if (row < nrows) out[(size_t)row * DFEAT + c] = acc[nt][rr];
        }
    }
}

// ---- message scatter: agg[dst] += norm * t[src]  (1 wave/edge, 2 f32/lane) ---
__global__ void scatter_k(const int* __restrict__ src, const int* __restrict__ dst,
                          const float* __restrict__ norm, const float* __restrict__ t,
                          float* __restrict__ agg, int E) {
    int e = blockIdx.x * 8 + (threadIdx.x >> 5);
    e = __builtin_amdgcn_readfirstlane(e);          // wave-uniform -> scalar loads
    if (e >= E) return;
    int lane = threadIdx.x & 31;
    int fp = lane * 2;
    int s = src[e], d = dst[e];
    float w = norm[e];
    float2 v = *(const float2*)(t + (size_t)s * DFEAT + fp);
    float* ad = agg + (size_t)d * DFEAT + fp;
    atomicAdd(ad,     w * v.x);
    atomicAdd(ad + 1, w * v.y);
}

// ---- epilogue: h = tanh(agg + dinv^2 * t + b)  (self-loop folded in) ---------
__global__ void epilogue_k(const float* __restrict__ agg, const float* __restrict__ t,
                           const float* __restrict__ dinv, const float* __restrict__ b,
                           float* __restrict__ h, int n) {
    long long i = (long long)blockIdx.x * blockDim.x + threadIdx.x;
    if (i >= (long long)n * DFEAT) return;
    int node = (int)(i >> 6);
    int f    = (int)(i & 63);
    float di = dinv[node];
    h[i] = tanhf(agg[i] + di * di * t[i] + b[f]);
}

// ---- pooling -----------------------------------------------------------------
__device__ __forceinline__ void atomicMaxF(float* addr, float v) {
    // valid for all non-NaN floats: int-max for >=0, uint-min for <0
    if (v >= 0.0f) atomicMax((int*)addr, __float_as_int(v));
    else           atomicMin((unsigned int*)addr, __float_as_uint(v));
}

__global__ void pool_k(const float* __restrict__ h, const int* __restrict__ batch,
                       float* __restrict__ maxb, float* __restrict__ sumb,
                       float* __restrict__ cnt, int n) {
    long long i = (long long)blockIdx.x * blockDim.x + threadIdx.x;
    if (i >= (long long)n * DFEAT) return;
    int node = (int)(i >> 6);
    int f    = (int)(i & 63);
    int g = batch[node];
    float v = h[i];
    atomicMaxF(&maxb[(size_t)g * DFEAT + f], v);
    atomicAdd(&sumb[(size_t)g * DFEAT + f], v);
    if (f == 0) atomicAdd(&cnt[g], 1.0f);
}

// ---- finalize: hidden = [max|mean], out = sigmoid(hidden . Wout + bout) ------
__global__ void finalize_k(const float* __restrict__ maxb, const float* __restrict__ sumb,
                           const float* __restrict__ cnt, const float* __restrict__ Wout,
                           const float* __restrict__ bout,
                           float* __restrict__ outp, float* __restrict__ hid, int G) {
    __shared__ float red[128];
    int g = blockIdx.x;
    int f = threadIdx.x;  // 0..127
    float c = cnt[g];
    float v;
    if (f < DFEAT) v = (c > 0.0f) ? maxb[(size_t)g * DFEAT + f] : 0.0f;
    else           v = sumb[(size_t)g * DFEAT + (f - DFEAT)] / fmaxf(c, 1.0f);
    hid[(size_t)g * 2 * DFEAT + f] = v;
    red[f] = v * Wout[f];
    __syncthreads();
    for (int st = 64; st > 0; st >>= 1) {
        if (f < st) red[f] += red[f + st];
        __syncthreads();
    }
    if (f == 0) {
        float z = red[0] + bout[0];
        outp[g] = 1.0f / (1.0f + expf(-z));
    }
}

// ---- host-side orchestration -------------------------------------------------
static inline unsigned nblk(long long n, int b) { return (unsigned)((n + b - 1) / b); }

extern "C" void kernel_launch(void* const* d_in, const int* in_sizes, int n_in,
                              void* d_out, int out_size, void* d_ws, size_t ws_size,
                              hipStream_t stream) {
    const float* x     = (const float*)d_in[0];
    const int*   eidx  = (const int*)d_in[1];
    const int*   batch = (const int*)d_in[2];
    const float* W[4]  = {(const float*)d_in[4], (const float*)d_in[6],
                          (const float*)d_in[8], (const float*)d_in[10]};
    const float* bb[4] = {(const float*)d_in[5], (const float*)d_in[7],
                          (const float*)d_in[9], (const float*)d_in[11]};
    const float* Wout  = (const float*)d_in[12];
    const float* bout  = (const float*)d_in[13];

    int N = in_sizes[2];
    int F = in_sizes[0] / N;
    int E = in_sizes[1] / 2;
    int G = out_size / (2 * DFEAT + 1);
    const int* src = eidx;
    const int* dst = eidx + E;

    // workspace layout (base assumed >=256B aligned)
    char* base = (char*)d_ws;
    __bf16* WT  = (__bf16*)base;                     // 64*128 bf16 max = 16KB
    float* dinv = (float*)(base + 16384);            // N
    float* norm = dinv + N;                          // E
    size_t ND = (size_t)N * DFEAT;
    float* h    = norm + E;                          // N*64
    float* t    = h + ND;                            // N*64
    float* agg  = t + ND;                            // N*64
    float* maxb = agg + ND;                          // G*64
    float* sumb = maxb + (size_t)G * DFEAT;          // G*64
    float* cnt  = sumb + (size_t)G * DFEAT;          // G
    (void)ws_size; (void)n_in;

    // degree normalization (self loop folded into init deg=1)
    fill_f32<<<nblk(N, 256), 256, 0, stream>>>(dinv, 1.0f, N);
    deg_accum<<<nblk(E, 256), 256, 0, stream>>>(dst, dinv, E);
    dinv_k<<<nblk(N, 256), 256, 0, stream>>>(dinv, N);
    norm_k<<<nblk(E, 256), 256, 0, stream>>>(src, dst, dinv, norm, E);

    for (int L = 0; L < 4; ++L) {
        const float* A = (L == 0) ? x : h;
        int K = (L == 0) ? F : DFEAT;
        wt_conv<<<nblk(DFEAT * K, 256), 256, 0, stream>>>(W[L], WT, K);
        if (K == 128)
            gemm_wmma_bf16<128><<<nblk(N, 128), 256, 0, stream>>>(A, WT, t, N);
        else
            gemm_wmma_bf16<64><<<nblk(N, 128), 256, 0, stream>>>(A, WT, t, N);
        fill_f32<<<nblk((long long)ND, 256), 256, 0, stream>>>(agg, 0.0f, (long long)ND);
        scatter_k<<<nblk(E, 8), 256, 0, stream>>>(src, dst, norm, t, agg, E);
        epilogue_k<<<nblk((long long)ND, 256), 256, 0, stream>>>(agg, t, dinv, bb[L], h, N);
    }

    // pooling + head
    fill_f32<<<nblk((long long)G * DFEAT, 256), 256, 0, stream>>>(maxb, -1e30f, (long long)G * DFEAT);
    fill_f32<<<nblk((long long)G * DFEAT, 256), 256, 0, stream>>>(sumb, 0.0f, (long long)G * DFEAT);
    fill_f32<<<nblk(G, 256), 256, 0, stream>>>(cnt, 0.0f, G);
    pool_k<<<nblk((long long)ND, 256), 256, 0, stream>>>(h, batch, maxb, sumb, cnt, N);

    float* outp = (float*)d_out;       // [G]
    float* hid  = outp + G;            // [G, 128]
    finalize_k<<<G, 128, 0, stream>>>(maxb, sumb, cnt, Wout, bout, outp, hid, G);
}
